// Speaker_70531952935219
// MI455X (gfx1250) — matrix operand order repaired and checked by reading, not compiled, not run on hardware
//
#include <hip/hip_runtime.h>

// MI455X / gfx1250, wave32. All GEMMs via v_wmma_f32_16x16x32_bf16.
// Weights converted once per call to bf16 in workspace (w_voc 32MB + embT 16MB
// + GRU weights ~5MB all resident in the 192MB L2 across the 64 steps).
// Each wave computes a 32x32 output tile: 4 accumulators, 4 WMMAs per K-chunk,
// halving A-fragment traffic vs a 32x16 tile. Streaming B operand is prefetched
// (global_prefetch_b8).

#define DEV __device__ __forceinline__

constexpr int BB = 32;     // batch
constexpr int SS = 64;     // steps
constexpr int VV = 32000;  // vocab
constexpr int DE = 256;    // embed dim
constexpr int DH = 512;    // hidden dim
constexpr int G3 = 3 * DH; // 1536 gate width

typedef __bf16 v16bf __attribute__((ext_vector_type(16)));
typedef float  v8f   __attribute__((ext_vector_type(8)));

union FragBF { v16bf v; uint4 q[2]; };

DEV unsigned short f2bf(float f) {
    unsigned u = __float_as_uint(f);
    unsigned r = u + 0x7FFFu + ((u >> 16) & 1u);   // round-to-nearest-even
    return (unsigned short)(r >> 16);
}
DEV float bf2f(unsigned short h) {
    return __uint_as_float(((unsigned)h) << 16);
}
DEV unsigned ford(float f) {               // monotonic float->uint
    unsigned u = __float_as_uint(f);
    return u ^ ((u & 0x80000000u) ? 0xFFFFFFFFu : 0x80000000u);
}
DEV float ford_inv(unsigned u) {
    u ^= ((u & 0x80000000u) ? 0x80000000u : 0xFFFFFFFFu);
    return __uint_as_float(u);
}
DEV float sigm(float x) { return 1.0f / (1.0f + __expf(-x)); }

DEV float gumbel_ln(int s, int b, int v) { // ln = log(Exp(1) sample)
    unsigned x = (unsigned)s * 0x9E3779B1u ^ (unsigned)b * 0x85EBCA6Bu ^
                 (unsigned)v * 0xC2B2AE35u;
    x ^= x >> 16; x *= 0x7FEB352Du; x ^= x >> 15; x *= 0x846CA68Bu; x ^= x >> 16;
    float u = ((float)x + 0.5f) * (1.0f / 4294967296.0f); // (0,1)
    return __logf(-__logf(u));
}

// ---------------------------------------------------------------------------
// Generic WMMA GEMM: D(32 x N) = A(32 x K) @ B(N x K)^T (+bias), bf16->f32.
// One wave owns a 32x32 tile (2 M-tiles x 2 N-tiles, 4 WMMAs per K-chunk).
// Block = 256 threads = 8 waves = 256 columns. grid.x = N/256.
// grid.y > 1 => split-K: kLen per chunk, f32 atomicAdd reduction (D pre-zeroed),
// bias added by chunk 0 only.
// ---------------------------------------------------------------------------
__global__ void wmma_gemm_a32bt(const unsigned short* __restrict__ A,
                                const unsigned short* __restrict__ B,
                                const float* __restrict__ bias,
                                float* __restrict__ D,
                                int N, int K, int kLen) {
    const int lane  = threadIdx.x & 31;
    const int wv    = threadIdx.x >> 5;
    const int hi    = lane >> 4;     // half-wave selects K sub-range
    const int l15   = lane & 15;
    const int cBase = blockIdx.x * 256 + wv * 32;
    const int kBase = blockIdx.y * kLen;
    const int kEnd  = kBase + kLen;

    const unsigned short* Ar0 = A + (size_t)l15 * K;           // M rows 0..15
    const unsigned short* Ar1 = A + (size_t)(l15 + 16) * K;    // M rows 16..31
    const unsigned short* Bc0 = B + (size_t)(cBase + l15) * K;        // N cols 0..15
    const unsigned short* Bc1 = B + (size_t)(cBase + 16 + l15) * K;   // N cols 16..31

    v8f acc00 = {}, acc10 = {}, acc01 = {}, acc11 = {};
    for (int k0 = kBase; k0 < kEnd; k0 += 32) {
        FragBF a0, a1, b0, b1;
        // A 16x32 bf16 frag: lane l15=row; hi picks K {0..7,16..23} vs {8..15,24..31}
        a0.q[0] = *(const uint4*)(Ar0 + k0 + hi * 8);
        a0.q[1] = *(const uint4*)(Ar0 + k0 + 16 + hi * 8);
        a1.q[0] = *(const uint4*)(Ar1 + k0 + hi * 8);
        a1.q[1] = *(const uint4*)(Ar1 + k0 + 16 + hi * 8);
        // B 32x16 bf16 frags: lane l15=column; hi picks K {0..7,16..23} vs {8..15,24..31}
        b0.q[0] = *(const uint4*)(Bc0 + k0 + hi * 8);
        b0.q[1] = *(const uint4*)(Bc0 + k0 + 16 + hi * 8);
        b1.q[0] = *(const uint4*)(Bc1 + k0 + hi * 8);
        b1.q[1] = *(const uint4*)(Bc1 + k0 + 16 + hi * 8);
        // pull the streaming B operand ahead (global_prefetch_b8)
        __builtin_prefetch(Bc0 + k0 + 512, 0, 0);
        __builtin_prefetch(Bc1 + k0 + 512, 0, 0);
        acc00 = __builtin_amdgcn_wmma_f32_16x16x32_bf16(false, a0.v, false, b0.v,
                                                        (short)0, acc00, false, false);
        acc10 = __builtin_amdgcn_wmma_f32_16x16x32_bf16(false, a1.v, false, b0.v,
                                                        (short)0, acc10, false, false);
        acc01 = __builtin_amdgcn_wmma_f32_16x16x32_bf16(false, a0.v, false, b1.v,
                                                        (short)0, acc01, false, false);
        acc11 = __builtin_amdgcn_wmma_f32_16x16x32_bf16(false, a1.v, false, b1.v,
                                                        (short)0, acc11, false, false);
    }

    const bool split = (gridDim.y > 1);
    const int  c0 = cBase + l15;
    const int  c1 = cBase + 16 + l15;
    const bool addB = (bias && (!split || blockIdx.y == 0));
    const float bv0 = addB ? bias[c0] : 0.0f;
    const float bv1 = addB ? bias[c1] : 0.0f;
#pragma unroll
    for (int r = 0; r < 8; r++) {
        int m = hi * 8 + r;                       // C/D layout: vgpr r -> M = r + 8*hi
        float v00 = acc00[r] + bv0;
        float v10 = acc10[r] + bv0;
        float v01 = acc01[r] + bv1;
        float v11 = acc11[r] + bv1;
        if (split) {
            atomicAdd(&D[(size_t)m * N + c0], v00);
            atomicAdd(&D[(size_t)(m + 16) * N + c0], v10);
            atomicAdd(&D[(size_t)m * N + c1], v01);
            atomicAdd(&D[(size_t)(m + 16) * N + c1], v11);
        } else {
            D[(size_t)m * N + c0] = v00;
            D[(size_t)(m + 16) * N + c0] = v10;
            D[(size_t)m * N + c1] = v01;
            D[(size_t)(m + 16) * N + c1] = v11;
        }
    }
}

// ---------------------------------------------------------------------------
// Elementwise / reduction kernels
// ---------------------------------------------------------------------------
__global__ void conv_bf16(const float* __restrict__ src,
                          unsigned short* __restrict__ dst, int n) {
    for (int i = blockIdx.x * blockDim.x + threadIdx.x; i < n;
         i += gridDim.x * blockDim.x)
        dst[i] = f2bf(src[i]);
}

__global__ void conv_embT(const float* __restrict__ emb,
                          unsigned short* __restrict__ dst) { // dst[d*V+v]=emb[v*DE+d]
    const int n = DE * VV;
    for (int i = blockIdx.x * blockDim.x + threadIdx.x; i < n;
         i += gridDim.x * blockDim.x) {
        int d = i / VV, v = i - d * VV;
        dst[i] = f2bf(emb[(size_t)v * DE + d]);
    }
}

__global__ void bcast_x0(const float* __restrict__ emb,
                         unsigned short* __restrict__ x0bf) { // BOS token id=2
    int i = blockIdx.x * 256 + threadIdx.x;        // B*DE = 8192
    if (i < BB * DE) x0bf[i] = f2bf(emb[2 * DE + (i & (DE - 1))]);
}

__global__ void prep_step(unsigned* rowmax, float* rowsum,
                          unsigned long long* amax, float* xzero) {
    int i = blockIdx.x * 256 + threadIdx.x;
    if (i < BB) { rowmax[i] = 0u; rowsum[i] = 0.0f; amax[i] = 0ull; }
    if (i < BB * DE) xzero[i] = 0.0f;              // split-K accumulator target
}

__global__ void noise_mask_rowmax(float* __restrict__ y,
                                  unsigned* __restrict__ rowmax, int step) {
    const int b = blockIdx.y;
    const int v = blockIdx.x * 256 + threadIdx.x;
    float t = y[(size_t)b * VV + v] - gumbel_ln(step, b, v);
    if (step == 0 && v == 3) t = -3.0e38f;         // mask EOS at step 0
    y[(size_t)b * VV + v] = t;
    __shared__ float sm[256];
    sm[threadIdx.x] = t; __syncthreads();
    for (int o = 128; o > 0; o >>= 1) {
        if (threadIdx.x < o) sm[threadIdx.x] = fmaxf(sm[threadIdx.x], sm[threadIdx.x + o]);
        __syncthreads();
    }
    if (threadIdx.x == 0) atomicMax(&rowmax[b], ford(sm[0]));
}

__global__ void exp_sum_argmax(const float* __restrict__ y,
                               const unsigned* __restrict__ rowmax,
                               float* __restrict__ rowsum,
                               unsigned long long* __restrict__ amax,
                               unsigned short* __restrict__ ebuf) {
    const int b = blockIdx.y;
    const int v = blockIdx.x * 256 + threadIdx.x;
    const float mx = ford_inv(rowmax[b]);
    const float t = y[(size_t)b * VV + v];
    const float e = __expf(t - mx);
    ebuf[(size_t)b * VV + v] = f2bf(e);
    // first-index argmax: bigger value wins, ties -> smaller v wins
    unsigned long long pk = ((unsigned long long)ford(t) << 32) |
                            (unsigned long long)(0xFFFFFFFFu - (unsigned)v);
    __shared__ float ss[256];
    __shared__ unsigned long long sp[256];
    ss[threadIdx.x] = e; sp[threadIdx.x] = pk; __syncthreads();
    for (int o = 128; o > 0; o >>= 1) {
        if (threadIdx.x < o) {
            ss[threadIdx.x] += ss[threadIdx.x + o];
            sp[threadIdx.x] = max(sp[threadIdx.x], sp[threadIdx.x + o]);
        }
        __syncthreads();
    }
    if (threadIdx.x == 0) {
        atomicAdd(&rowsum[b], ss[0]);
        atomicMax(&amax[b], sp[0]);
    }
}

__global__ void finalize_step(unsigned short* __restrict__ ebuf,
                              const float* __restrict__ rowsum,
                              const unsigned long long* __restrict__ amax,
                              float* __restrict__ out_logits,
                              int* __restrict__ out_labels, int step) {
    const int b = blockIdx.y;
    const int v = blockIdx.x * 256 + threadIdx.x;
    const float inv = 1.0f / rowsum[b];
    const float c = bf2f(ebuf[(size_t)b * VV + v]) * inv;
    out_logits[((size_t)b * SS + step) * VV + v] = c;   // y_soft -> output
    ebuf[(size_t)b * VV + v] = f2bf(c);                 // normalized A for emb GEMM
    if (v == 0)
        out_labels[b * SS + step] =
            (int)(0xFFFFFFFFu - (unsigned)(amax[b] & 0xFFFFFFFFull));
}

__global__ void gru_gate(const float* __restrict__ gi, const float* __restrict__ gh,
                         const float* __restrict__ hin, float* __restrict__ hout,
                         unsigned short* __restrict__ hout_bf) {
    int i = blockIdx.x * 256 + threadIdx.x;        // B*DH = 16384
    if (i >= BB * DH) return;
    int b = i >> 9, j = i & (DH - 1);
    const float* gib = gi + b * G3;
    const float* ghb = gh + b * G3;
    float r = sigm(gib[j] + ghb[j]);
    float z = sigm(gib[DH + j] + ghb[DH + j]);
    float n = tanhf(gib[2 * DH + j] + r * ghb[2 * DH + j]);
    float h = (1.0f - z) * n + z * hin[i];
    hout[i] = h;
    hout_bf[i] = f2bf(h);
}

__global__ void post_labels(int* __restrict__ labels, int* __restrict__ caplen,
                            int* __restrict__ pbuf) {
    int b = threadIdx.x;
    if (b >= BB) return;
    int* lb = labels + b * SS;
    lb[SS - 1] = 3;
    int p = SS - 1;
    for (int t = 0; t < SS; t++) if (lb[t] == 3) { p = t; break; }
    for (int t = p; t < SS; t++) lb[t] = 0;        // labels * pad_g
    caplen[b] = p + 1;
    pbuf[b] = p;
}

__global__ void post_logits(float* __restrict__ logits, const int* __restrict__ pbuf) {
    const int v = blockIdx.x * 256 + threadIdx.x;
    const int t = blockIdx.y, b = blockIdx.z;
    if (t >= pbuf[b])
        logits[((size_t)b * SS + t) * VV + v] = (v == 0) ? 1.0f : 0.0f;
}

// ---------------------------------------------------------------------------
// Host orchestration (all launches on `stream`; graph-capture safe)
// ---------------------------------------------------------------------------
extern "C" void kernel_launch(void* const* d_in, const int* in_sizes, int n_in,
                              void* d_out, int out_size, void* d_ws, size_t ws_size,
                              hipStream_t stream) {
    const float* h_img = (const float*)d_in[0];
    const float* emb   = (const float*)d_in[1];
    const float* w_ih0 = (const float*)d_in[2];
    const float* w_hh0 = (const float*)d_in[3];
    const float* b_ih0 = (const float*)d_in[4];
    const float* b_hh0 = (const float*)d_in[5];
    const float* w_ih1 = (const float*)d_in[6];
    const float* w_hh1 = (const float*)d_in[7];
    const float* b_ih1 = (const float*)d_in[8];
    const float* b_hh1 = (const float*)d_in[9];
    const float* w_voc = (const float*)d_in[10];
    const float* b_voc = (const float*)d_in[11];

    // workspace bump allocator (256B aligned); total ~61 MB
    char* ws = (char*)d_ws;
    size_t off = 0;
    auto alloc = [&](size_t bytes) -> void* {
        void* p = ws + off;
        off = (off + bytes + 255) & ~(size_t)255;
        return p;
    };
    unsigned short* wvoc_bf = (unsigned short*)alloc((size_t)VV * DH * 2);
    unsigned short* embT    = (unsigned short*)alloc((size_t)DE * VV * 2);
    unsigned short* wih0    = (unsigned short*)alloc((size_t)G3 * DE * 2);
    unsigned short* whh0    = (unsigned short*)alloc((size_t)G3 * DH * 2);
    unsigned short* wih1    = (unsigned short*)alloc((size_t)G3 * DH * 2);
    unsigned short* whh1    = (unsigned short*)alloc((size_t)G3 * DH * 2);
    unsigned short* himg_bf = (unsigned short*)alloc((size_t)BB * DH * 2);
    unsigned short* x_bf    = (unsigned short*)alloc((size_t)BB * DE * 2);
    float*          x_f     = (float*)alloc((size_t)BB * DE * 4);
    float*          h0_f    = (float*)alloc((size_t)BB * DH * 4);
    unsigned short* h0_bf   = (unsigned short*)alloc((size_t)BB * DH * 2);
    float*          h1_f    = (float*)alloc((size_t)BB * DH * 4);
    unsigned short* h1_bf   = (unsigned short*)alloc((size_t)BB * DH * 2);
    float*          y       = (float*)alloc((size_t)BB * VV * 4);
    unsigned short* ebuf    = (unsigned short*)alloc((size_t)BB * VV * 2);
    float*          gi      = (float*)alloc((size_t)BB * G3 * 4);
    float*          gh      = (float*)alloc((size_t)BB * G3 * 4);
    unsigned*       rowmax  = (unsigned*)alloc(BB * 4);
    float*          rowsum  = (float*)alloc(BB * 4);
    unsigned long long* amax = (unsigned long long*)alloc(BB * 8);
    int*            pbuf    = (int*)alloc(BB * 4);

    float* out_logits = (float*)d_out;                       // (B,S,V) f32
    int*   out_labels = (int*)(out_logits + (size_t)BB * SS * VV); // (B,S) i32
    int*   out_cap    = out_labels + BB * SS;                // (B,) i32

    // ---- one-time bf16 weight staging (then L2-resident for all 64 steps) ----
    conv_bf16<<<2048, 256, 0, stream>>>(w_voc, wvoc_bf, VV * DH);
    conv_embT<<<2048, 256, 0, stream>>>(emb, embT);
    conv_bf16<<<512, 256, 0, stream>>>(w_ih0, wih0, G3 * DE);
    conv_bf16<<<512, 256, 0, stream>>>(w_hh0, whh0, G3 * DH);
    conv_bf16<<<512, 256, 0, stream>>>(w_ih1, wih1, G3 * DH);
    conv_bf16<<<512, 256, 0, stream>>>(w_hh1, whh1, G3 * DH);
    conv_bf16<<<64, 256, 0, stream>>>(h_img, himg_bf, BB * DH);
    bcast_x0<<<32, 256, 0, stream>>>(emb, x_bf);

    // ---- initial GRU step on BOS embedding, init state = h_img per layer ----
    wmma_gemm_a32bt<<<dim3(G3 / 256, 1), 256, 0, stream>>>(x_bf, wih0, b_ih0, gi, G3, DE, DE);
    wmma_gemm_a32bt<<<dim3(G3 / 256, 1), 256, 0, stream>>>(himg_bf, whh0, b_hh0, gh, G3, DH, DH);
    gru_gate<<<64, 256, 0, stream>>>(gi, gh, h_img, h0_f, h0_bf);
    wmma_gemm_a32bt<<<dim3(G3 / 256, 1), 256, 0, stream>>>(h0_bf, wih1, b_ih1, gi, G3, DH, DH);
    wmma_gemm_a32bt<<<dim3(G3 / 256, 1), 256, 0, stream>>>(himg_bf, whh1, b_hh1, gh, G3, DH, DH);
    gru_gate<<<64, 256, 0, stream>>>(gi, gh, h_img, h1_f, h1_bf);

    // ---- 64 sequential decode steps ----
    for (int s = 0; s < SS; s++) {
        prep_step<<<32, 256, 0, stream>>>(rowmax, rowsum, amax, x_f);
        // logits = h1 @ w_voc^T + b_voc : 125 WGs * 8 waves = 1000 32x32 tiles
        wmma_gemm_a32bt<<<dim3(VV / 256, 1), 256, 0, stream>>>(h1_bf, wvoc_bf, b_voc, y, VV, DH, DH);
        noise_mask_rowmax<<<dim3(VV / 256, BB), 256, 0, stream>>>(y, rowmax, s);
        exp_sum_argmax<<<dim3(VV / 256, BB), 256, 0, stream>>>(y, rowmax, rowsum, amax, ebuf);
        finalize_step<<<dim3(VV / 256, BB), 256, 0, stream>>>(ebuf, rowsum, amax,
                                                              out_logits, out_labels, s);
        // x = c @ emb : split-K (10 chunks of 3200) into zeroed x_f
        wmma_gemm_a32bt<<<dim3(DE / 256, 10), 256, 0, stream>>>(ebuf, embT, nullptr, x_f, DE, VV, VV / 10);
        conv_bf16<<<32, 256, 0, stream>>>(x_f, x_bf, BB * DE);
        // GRU layer 0
        wmma_gemm_a32bt<<<dim3(G3 / 256, 1), 256, 0, stream>>>(x_bf, wih0, b_ih0, gi, G3, DE, DE);
        wmma_gemm_a32bt<<<dim3(G3 / 256, 1), 256, 0, stream>>>(h0_bf, whh0, b_hh0, gh, G3, DH, DH);
        gru_gate<<<64, 256, 0, stream>>>(gi, gh, h0_f, h0_f, h0_bf);
        // GRU layer 1
        wmma_gemm_a32bt<<<dim3(G3 / 256, 1), 256, 0, stream>>>(h0_bf, wih1, b_ih1, gi, G3, DH, DH);
        wmma_gemm_a32bt<<<dim3(G3 / 256, 1), 256, 0, stream>>>(h1_bf, whh1, b_hh1, gh, G3, DH, DH);
        gru_gate<<<64, 256, 0, stream>>>(gi, gh, h1_f, h1_f, h1_bf);
    }

    // ---- EOS masking / cap_len postprocess ----
    post_labels<<<1, 32, 0, stream>>>(out_labels, out_cap, pbuf);
    post_logits<<<dim3(VV / 256, SS, BB), 256, 0, stream>>>(out_logits, pbuf);
}
